// ResidualDDFBlock_70248485094082
// MI455X (gfx1250) — matrix operand order repaired and checked by reading, not compiled, not run on hardware
//
#include <hip/hip_runtime.h>
#include <math.h>

typedef __attribute__((ext_vector_type(16))) _Float16 v16h;
typedef __attribute__((ext_vector_type(8)))  float    v8f;

#define NF   64
#define MID  12
#define HW   256
#define TILE 16
#define HALO 18
#define CSTRIDE (HALO*HALO)     // 324 floats per channel plane in LDS
#define PLANE (HW*HW)           // 65536

#if defined(__HIP_DEVICE_COMPILE__) && __has_builtin(__builtin_amdgcn_global_load_async_to_lds_b32)
#define HAVE_ASYNC_LDS 1
#else
#define HAVE_ASYNC_LDS 0
#endif

// ---------------------------------------------------------------------------
// Global average pool: one block per (b,c) plane; g[bc] = mean(plane)
// ---------------------------------------------------------------------------
__global__ __launch_bounds__(256) void gap_kernel(const float* __restrict__ x,
                                                  float* __restrict__ g) {
  __shared__ float red[256];
  const int bc = blockIdx.x;
  const float* p = x + (size_t)bc * PLANE;
  float s = 0.f;
  for (int i = threadIdx.x; i < PLANE; i += 256) s += p[i];
  red[threadIdx.x] = s;
  __syncthreads();
  for (int off = 128; off > 0; off >>= 1) {
    if (threadIdx.x < off) red[threadIdx.x] += red[threadIdx.x + off];
    __syncthreads();
  }
  if (threadIdx.x == 0) g[bc] = red[0] * (1.f / (float)PLANE);
}

// ---------------------------------------------------------------------------
// SE-MLP (64 -> 12 relu -> 576) + per-channel 9-tap filter norm -> cf (4,64,9)
// ---------------------------------------------------------------------------
__global__ __launch_bounds__(256) void mlp_kernel(const float* __restrict__ g,
                                                  const float* __restrict__ cw1,
                                                  const float* __restrict__ cb1,
                                                  const float* __restrict__ cw2,
                                                  const float* __restrict__ cb2,
                                                  const float* __restrict__ cstd,
                                                  float* __restrict__ cf) {
  __shared__ float hs[4 * MID];
  __shared__ float cfr[4 * NF * 9];
  const int t = threadIdx.x;
  if (t < 4 * MID) {
    const int b = t / MID, m = t % MID;
    float acc = cb1[m];
    for (int c = 0; c < NF; ++c) acc += g[b * NF + c] * cw1[m * NF + c];
    hs[t] = fmaxf(acc, 0.f);
  }
  __syncthreads();
  for (int idx = t; idx < 4 * NF * 9; idx += 256) {
    const int b = idx / (NF * 9), o = idx % (NF * 9);
    float acc = cb2[o];
    for (int m = 0; m < MID; ++m) acc += hs[b * MID + m] * cw2[o * MID + m];
    cfr[idx] = acc;
  }
  __syncthreads();
  {  // 256 threads == 4*64 (b,c) pairs: normalize the 9 taps
    const int b = t / NF, c = t % NF;
    float v[9], sum = 0.f;
    for (int k = 0; k < 9; ++k) { v[k] = cfr[b * NF * 9 + c * 9 + k]; sum += v[k]; }
    const float mean = sum * (1.f / 9.f);
    float var = 0.f;
    for (int k = 0; k < 9; ++k) { const float d = v[k] - mean; var += d * d; }
    var *= (1.f / 8.f);  // unbiased (ddof=1)
    const float inv = 1.f / (sqrtf(var) + 1e-10f);
    for (int k = 0; k < 9; ++k)
      cf[b * NF * 9 + c * 9 + k] = (v[k] - mean) * inv * cstd[c * 9 + k];
  }
}

// ---------------------------------------------------------------------------
// Fused DDF pack: 1x1-conv spatial filter via WMMA + tap-norm + 3x3 apply.
// MODE 0: out = relu(ddf)   MODE 1: out = ddf + resid
// ---------------------------------------------------------------------------
template <int MODE>
__global__ __launch_bounds__(256) void ddf_kernel(const float* __restrict__ x,
                                                  const float* __restrict__ sw,
                                                  const float* __restrict__ sb,
                                                  const float* __restrict__ sstd,
                                                  const float* __restrict__ cf,
                                                  const float* __restrict__ resid,
                                                  float* __restrict__ out) {
  extern __shared__ float smem[];
  float* xs = smem;                 // [64][18][18]  (channel plane stride 324)
  float* ss = smem + NF * CSTRIDE;  // [256][9] spatial filter

  const int t   = threadIdx.x;
  const int b   = blockIdx.z;
  const int ty0 = blockIdx.y * TILE;
  const int tx0 = blockIdx.x * TILE;

  // --- stage haloed fp32 tile into LDS (zero-padded borders) ---
  for (int idx = t; idx < NF * CSTRIDE; idx += 256) {
    const int c  = idx / CSTRIDE;
    const int r  = idx - c * CSTRIDE;
    const int ly = r / HALO, lx = r - ly * HALO;
    const int gy = ty0 - 1 + ly, gx = tx0 - 1 + lx;
    if ((unsigned)gy < (unsigned)HW && (unsigned)gx < (unsigned)HW) {
      const size_t gi = (((size_t)b * NF + c) * HW + gy) * HW + gx;
#if HAVE_ASYNC_LDS
      __builtin_amdgcn_global_load_async_to_lds_b32(
          (__attribute__((address_space(1))) int*)&x[gi],
          (__attribute__((address_space(3))) int*)&xs[idx], 0, 0);
#else
      xs[idx] = x[gi];
#endif
    } else {
      xs[idx] = 0.f;  // zero-pad halo
    }
  }
#if HAVE_ASYNC_LDS
#if __has_builtin(__builtin_amdgcn_s_wait_asynccnt)
  __builtin_amdgcn_s_wait_asynccnt(0);
#else
  asm volatile("s_wait_asynccnt 0x0" ::: "memory");
#endif
#endif
  __syncthreads();

  // --- spatial 1x1 conv via WMMA f32 = f16 x f16 (K=64 split as 2x32) ---
  {
    const int wave = t >> 5;
    const int lane = t & 31;
    const int n    = lane & 15;       // tap column (0..8 valid, 9..15 zero-pad)
    const bool hi  = lane >= 16;

    // B fragments (32x16), built once per wave; B[k][n] = sw[n][k]
    v16h blo, bhi;
    const int kb2 = hi ? 8 : 0;
    for (int j = 0; j < 16; ++j) {
      const int k = kb2 + j + (j >= 8 ? 8 : 0);
      blo[j] = (_Float16)((n < 9) ? sw[n * NF + k]      : 0.f);
      bhi[j] = (_Float16)((n < 9) ? sw[n * NF + 32 + k] : 0.f);
    }
    const float cini = (n < 9) ? sb[n] : 0.f;  // bias folded into accumulator

    const int m  = lane & 15;         // pixel-in-row (A's M dim)
    const int kb = hi ? 8 : 0;
    for (int rr = 0; rr < 2; ++rr) {
      const int r    = wave + rr * 8;            // tile row (pixel group)
      const int base = (r + 1) * HALO + (m + 1); // interior pixel in halo coords
      v16h alo, ahi;
      for (int j = 0; j < 8; ++j) {
        const int ch0 = kb + j, ch1 = 16 + kb + j;
        alo[j]     = (_Float16)xs[ch0 * CSTRIDE + base];
        alo[8 + j] = (_Float16)xs[ch1 * CSTRIDE + base];
        ahi[j]     = (_Float16)xs[(32 + ch0) * CSTRIDE + base];
        ahi[8 + j] = (_Float16)xs[(32 + ch1) * CSTRIDE + base];
      }
      v8f acc;
      for (int v = 0; v < 8; ++v) acc[v] = cini;
      acc = __builtin_amdgcn_wmma_f32_16x16x32_f16(false, alo, false, blo,
                                                   (short)0, acc, false, false);
      acc = __builtin_amdgcn_wmma_f32_16x16x32_f16(false, ahi, false, bhi,
                                                   (short)0, acc, false, false);
      if (n < 9) {  // D: lane holds N=n, VGPR v holds M = v + 8*(lane>=16)
        const int mb = hi ? 8 : 0;
        for (int v = 0; v < 8; ++v)
          ss[(r * TILE + v + mb) * 9 + n] = acc[v];
      }
    }
  }
  __syncthreads();

  // --- normalize s over 9 taps (unbiased std) and scale by sstd ---
  {
    float v[9], sum = 0.f;
    for (int k = 0; k < 9; ++k) { v[k] = ss[t * 9 + k]; sum += v[k]; }
    const float mean = sum * (1.f / 9.f);
    float var = 0.f;
    for (int k = 0; k < 9; ++k) { const float d = v[k] - mean; var += d * d; }
    var *= (1.f / 8.f);
    const float inv = 1.f / (sqrtf(var) + 1e-10f);
    for (int k = 0; k < 9; ++k) ss[t * 9 + k] = (v[k] - mean) * inv * sstd[k];
  }
  __syncthreads();

  // --- apply decoupled dynamic filter; thread = pixel, loop channels ---
  {
    const int ty = t >> 4, tx = t & 15;
    float sk[9];
    #pragma unroll
    for (int k = 0; k < 9; ++k) sk[k] = ss[t * 9 + k];
    const float* __restrict__ cfg = cf + b * NF * 9;  // uniform -> SMEM loads
    const size_t obase = (((size_t)b * NF) * HW + (ty0 + ty)) * HW + (tx0 + tx);
    for (int c = 0; c < NF; ++c) {
      const float* xc = xs + c * CSTRIDE + ty * HALO + tx;
      float acc = 0.f;
      #pragma unroll
      for (int k = 0; k < 9; ++k)
        acc += cfg[c * 9 + k] * sk[k] * xc[(k / 3) * HALO + (k % 3)];
      const size_t oi = obase + (size_t)c * PLANE;
      out[oi] = (MODE == 0) ? fmaxf(acc, 0.f) : acc + resid[oi];
    }
  }
}

// ---------------------------------------------------------------------------
extern "C" void kernel_launch(void* const* d_in, const int* in_sizes, int n_in,
                              void* d_out, int out_size, void* d_ws, size_t ws_size,
                              hipStream_t stream) {
  const float* x     = (const float*)d_in[0];
  const float* sw1   = (const float*)d_in[1];
  const float* sb1   = (const float*)d_in[2];
  const float* sstd1 = (const float*)d_in[3];
  const float* cw1a  = (const float*)d_in[4];
  const float* cb1a  = (const float*)d_in[5];
  const float* cw2a  = (const float*)d_in[6];
  const float* cb2a  = (const float*)d_in[7];
  const float* cstd1 = (const float*)d_in[8];
  const float* sw2   = (const float*)d_in[9];
  const float* sb2   = (const float*)d_in[10];
  const float* sstd2 = (const float*)d_in[11];
  const float* cw1b  = (const float*)d_in[12];
  const float* cb1b  = (const float*)d_in[13];
  const float* cw2b  = (const float*)d_in[14];
  const float* cb2b  = (const float*)d_in[15];
  const float* cstd2 = (const float*)d_in[16];

  float* y1 = (float*)d_ws;                 // (4,64,256,256) intermediate
  float* g  = y1 + (size_t)4 * NF * PLANE;  // (4,64) GAP
  float* cf = g + 4 * NF;                   // (4,64,9) channel filter

  const dim3 grid(HW / TILE, HW / TILE, 4);
  const dim3 block(256);
  const size_t shmem = (size_t)(NF * CSTRIDE + 256 * 9) * sizeof(float);

  // pack 1: out1 = relu(ddf(x))
  gap_kernel<<<4 * NF, 256, 0, stream>>>(x, g);
  mlp_kernel<<<1, 256, 0, stream>>>(g, cw1a, cb1a, cw2a, cb2a, cstd1, cf);
  ddf_kernel<0><<<grid, block, shmem, stream>>>(x, sw1, sb1, sstd1, cf, x, y1);

  // pack 2: out = x + ddf(out1)
  gap_kernel<<<4 * NF, 256, 0, stream>>>(y1, g);
  mlp_kernel<<<1, 256, 0, stream>>>(g, cw1b, cb1b, cw2b, cb2b, cstd2, cf);
  ddf_kernel<1><<<grid, block, shmem, stream>>>(y1, sw2, sb2, sstd2, cf, x,
                                                (float*)d_out);
}
